// TransformerBlock_292057776892
// MI455X (gfx1250) — matrix operand order repaired and checked by reading, not compiled, not run on hardware
//
#include <hip/hip_runtime.h>

// ---------------------------------------------------------------------------
// Transformer block for MI455X (gfx1250), bf16 WMMA everywhere.
// N=2, L=2048, E=1024, H=16, HD=64, FF=4096.
// ---------------------------------------------------------------------------

typedef __attribute__((ext_vector_type(16))) __bf16 v16bf;
typedef __attribute__((ext_vector_type(8)))  float  v8f;

union Frag {
    v16bf    v;
    unsigned u[8];
};

__device__ __forceinline__ unsigned short f2bf(float f) {
    unsigned u = __float_as_uint(f);
    unsigned r = u + 0x7FFFu + ((u >> 16) & 1u);   // round-to-nearest-even
    return (unsigned short)(r >> 16);
}

__device__ __forceinline__ uint2 pack4bf(float4 v) {
    uint2 p;
    p.x = (unsigned)f2bf(v.x) | ((unsigned)f2bf(v.y) << 16);
    p.y = (unsigned)f2bf(v.z) | ((unsigned)f2bf(v.w) << 16);
    return p;
}

__device__ __forceinline__ v8f zero8() {
    v8f z;
#pragma unroll
    for (int i = 0; i < 8; ++i) z[i] = 0.0f;
    return z;
}

// A-matrix fragment (16xK=32, bf16). ISA layout: lanes 0-15 -> rows, halves
// 0..7 hold K {0..7}, halves 8..15 hold K {16..23}; lanes 16-31 get +8 on K.
// LDS layout required: row-major, K contiguous (ld elements per row).
__device__ __forceinline__ void load_a(const unsigned short* base, int ld, int koff,
                                       int lane, Frag& f) {
    const int m = lane & 15, g = lane >> 4;
    const unsigned short* p = base + m * ld + koff + 8 * g;
#pragma unroll
    for (int j = 0; j < 8; ++j) {
        const int k = 2 * j + (j >= 4 ? 8 : 0);
        f.u[j] = *(const unsigned*)(p + k);
    }
}

// B-matrix fragment (K=32 x 16, bf16). Lanes 0-15: halves 0..7 hold K {0..7},
// halves 8..15 hold K {16..23}; lanes 16-31 get +8 on K. LDS layout required: column(N)-major, K contiguous.
__device__ __forceinline__ void load_b(const unsigned short* base, int ld, int koff,
                                       int lane, Frag& f) {
    const int nn = lane & 15, g = lane >> 4;
    const unsigned short* p = base + nn * ld + koff + 8 * g;
#pragma unroll
    for (int j = 0; j < 8; ++j) f.u[j] = *(const unsigned*)(p + 2 * j + (j >= 4 ? 8 : 0));
}

__device__ __forceinline__ v8f wmma_bf16(const Frag& a, const Frag& b, v8f c) {
    return __builtin_amdgcn_wmma_f32_16x16x32_bf16(false, a.v, false, b.v,
                                                   (short)0, c, false, false);
}

// ---------------------------------------------------------------------------
// Flash attention: grid (L/64, N*H), 128 threads (4 waves).
// Each wave owns a 16-row Q stripe; K/V streamed in 64-wide chunks via LDS.
// ---------------------------------------------------------------------------
__global__ __launch_bounds__(128) void attn_kernel(const float* __restrict__ Q,
                                                   const float* __restrict__ K,
                                                   const float* __restrict__ V,
                                                   float* __restrict__ O) {
    __shared__ unsigned short kbuf[64 * 64];        // K chunk, row-major (kk,hd)
    __shared__ unsigned short vtbuf[64 * 64];       // V chunk, transposed (hd,kk)
    __shared__ unsigned short pbuf[4][16 * 64];     // per-wave P staging

    const int tid = threadIdx.x, lane = tid & 31, w = tid >> 5;
    const int qbase = blockIdx.x * 64;
    const int nh = blockIdx.y;
    const int n = nh >> 4, h = nh & 15;
    const size_t base = ((size_t)n * 2048) * 1024 + (size_t)h * 64; // +pos*1024+hd

    // ---- stage Q block (64x64) into LDS (reuse kbuf), build A-fragments ----
    for (int i = tid; i < 1024; i += 128) {          // 1024 float4s
        const int row = i >> 4, cq = (i & 15) * 4;
        const float4 v = *(const float4*)&Q[base + (size_t)(qbase + row) * 1024 + cq];
        *(uint2*)&kbuf[row * 64 + cq] = pack4bf(v);
    }
    __syncthreads();
    Frag qa0, qa1;
    load_a(&kbuf[(w * 16) * 64], 64, 0,  lane, qa0);
    load_a(&kbuf[(w * 16) * 64], 64, 32, lane, qa1);

    v8f acc[4];
#pragma unroll
    for (int d = 0; d < 4; ++d) acc[d] = zero8();
    float mrow[8], srow[8];
#pragma unroll
    for (int r = 0; r < 8; ++r) { mrow[r] = -3.0e38f; srow[r] = 0.0f; }

    const float sc = 0.03125f;   // 1/sqrt(E)

    for (int kk0 = 0; kk0 < 2048; kk0 += 64) {
        __syncthreads();
        for (int i = tid; i < 1024; i += 128) {
            const int row = i >> 4, cq = (i & 15) * 4;
            const size_t g = base + (size_t)(kk0 + row) * 1024 + cq;
            const float4 kv = *(const float4*)&K[g];
            const float4 vv = *(const float4*)&V[g];
            *(uint2*)&kbuf[row * 64 + cq] = pack4bf(kv);
            vtbuf[(cq + 0) * 64 + row] = f2bf(vv.x);
            vtbuf[(cq + 1) * 64 + row] = f2bf(vv.y);
            vtbuf[(cq + 2) * 64 + row] = f2bf(vv.z);
            vtbuf[(cq + 3) * 64 + row] = f2bf(vv.w);
        }
        if (kk0 + 64 < 2048) {   // hide next chunk's HBM latency behind WMMA
            const int row = tid >> 1, cq = (tid & 1) * 32;
            __builtin_prefetch(&K[base + (size_t)(kk0 + 64 + row) * 1024 + cq], 0, 1);
            __builtin_prefetch(&V[base + (size_t)(kk0 + 64 + row) * 1024 + cq], 0, 1);
        }
        __syncthreads();

        // ---- S = Q * K^T  (4 tiles of 16x16, K-dim = HD = 64) ----
        v8f s[4];
#pragma unroll
        for (int t = 0; t < 4; ++t) {
            Frag b0, b1;
            load_b(&kbuf[(t * 16) * 64], 64, 0,  lane, b0);
            load_b(&kbuf[(t * 16) * 64], 64, 32, lane, b1);
            v8f z = zero8();
            z = wmma_bf16(qa0, b0, z);
            z = wmma_bf16(qa1, b1, z);
            s[t] = z;
        }

        // ---- online softmax (rows live in VGPR index r within each half) ----
        float tmax[8], al[8], ps[8];
#pragma unroll
        for (int t = 0; t < 4; ++t)
#pragma unroll
            for (int r = 0; r < 8; ++r) s[t][r] = s[t][r] * sc;
#pragma unroll
        for (int r = 0; r < 8; ++r)
            tmax[r] = fmaxf(fmaxf(s[0][r], s[1][r]), fmaxf(s[2][r], s[3][r]));
#pragma unroll
        for (int r = 0; r < 8; ++r) {
#pragma unroll
            for (int m = 1; m < 16; m <<= 1)
                tmax[r] = fmaxf(tmax[r], __shfl_xor(tmax[r], m, 32));
            const float nm = fmaxf(mrow[r], tmax[r]);
            al[r] = __expf(mrow[r] - nm);
            mrow[r] = nm;
            ps[r] = 0.0f;
        }
#pragma unroll
        for (int t = 0; t < 4; ++t)
#pragma unroll
            for (int r = 0; r < 8; ++r) {
                const float p = __expf(s[t][r] - mrow[r]);
                s[t][r] = p;
                ps[r] += p;
            }
#pragma unroll
        for (int r = 0; r < 8; ++r) {
#pragma unroll
            for (int m = 1; m < 16; m <<= 1) ps[r] += __shfl_xor(ps[r], m, 32);
            srow[r] = srow[r] * al[r] + ps[r];
        }
#pragma unroll
        for (int d = 0; d < 4; ++d)
#pragma unroll
            for (int r = 0; r < 8; ++r) acc[d][r] = acc[d][r] * al[r];

        // ---- C-layout -> A-layout for P via per-wave LDS staging ----
        unsigned short* pb = pbuf[w];
#pragma unroll
        for (int t = 0; t < 4; ++t)
#pragma unroll
            for (int r = 0; r < 8; ++r)
                pb[(r + 8 * (lane >> 4)) * 64 + t * 16 + (lane & 15)] = f2bf(s[t][r]);

        Frag pa0, pa1;
        load_a(pb, 64, 0,  lane, pa0);
        load_a(pb, 64, 32, lane, pa1);

        // ---- O += P * V ----
#pragma unroll
        for (int d = 0; d < 4; ++d) {
            Frag vb0, vb1;
            load_b(&vtbuf[(d * 16) * 64], 64, 0,  lane, vb0);
            load_b(&vtbuf[(d * 16) * 64], 64, 32, lane, vb1);
            acc[d] = wmma_bf16(pa0, vb0, acc[d]);
            acc[d] = wmma_bf16(pa1, vb1, acc[d]);
        }
    }

    // ---- normalize & store ----
#pragma unroll
    for (int d = 0; d < 4; ++d)
#pragma unroll
        for (int r = 0; r < 8; ++r) {
            const int row = w * 16 + r + 8 * (lane >> 4);
            const int col = d * 16 + (lane & 15);
            O[base + (size_t)(qbase + row) * 1024 + col] = acc[d][r] / srow[r];
        }
}

// ---------------------------------------------------------------------------
// GEMM: C = act(A@B + bias [+ resid]). 128x128 tile, K-step 64, 256 threads.
// Each wave (of 8) computes a 32x64 sub-tile = 2x4 WMMA accumulators,
// 16 WMMAs per K-step between barriers.
// ---------------------------------------------------------------------------
__global__ __launch_bounds__(256) void gemm_kernel(const float* __restrict__ A,
                                                   const float* __restrict__ B,
                                                   const float* __restrict__ bias,
                                                   const float* __restrict__ resid,
                                                   float* __restrict__ Cout,
                                                   int M, int Nn, int Kd, int relu) {
    __shared__ unsigned short sa[128 * 64];   // row-major (m,k)   16KB
    __shared__ unsigned short sb[128 * 64];   // n-major   (n,k)   16KB

    const int tid = threadIdx.x, lane = tid & 31, w = tid >> 5;
    const int n0 = blockIdx.x * 128, m0 = blockIdx.y * 128;
    const int moff = (w >> 1) * 32, noff = (w & 1) * 64;

    v8f acc[2][4];
#pragma unroll
    for (int i = 0; i < 2; ++i)
#pragma unroll
        for (int j = 0; j < 4; ++j) acc[i][j] = zero8();

    const int am = tid >> 1, aq = (tid & 1) * 8;     // A fill: 8 float4/thread
    const int bk = tid >> 2, bq = (tid & 3) * 8;     // B fill: 8 float4/thread

    for (int k0 = 0; k0 < Kd; k0 += 64) {
        __syncthreads();
        // ---- A tile 128x64: float4 global loads, b64 packed LDS stores ----
        {
            const float4* ap = (const float4*)(A + (size_t)(m0 + am) * Kd + k0) + aq;
#pragma unroll
            for (int i = 0; i < 8; ++i)
                *(uint2*)&sa[am * 64 + (aq + i) * 4] = pack4bf(ap[i]);
        }
        // ---- B tile 64x128: float4 global loads, transposed LDS stores ----
        {
            const float4* bp = (const float4*)(B + (size_t)(k0 + bk) * Nn + n0) + bq;
#pragma unroll
            for (int i = 0; i < 8; ++i) {
                const float4 v = bp[i];
                const int c = (bq + i) * 4;
                sb[(c + 0) * 64 + bk] = f2bf(v.x);
                sb[(c + 1) * 64 + bk] = f2bf(v.y);
                sb[(c + 2) * 64 + bk] = f2bf(v.z);
                sb[(c + 3) * 64 + bk] = f2bf(v.w);
            }
        }
        if (k0 + 64 < Kd) {   // prefetch next K-tile into L2/WGP$
            __builtin_prefetch(&A[(size_t)(m0 + am) * Kd + k0 + 64 + aq * 4], 0, 1);
            __builtin_prefetch(&B[(size_t)(k0 + 64 + bk) * Nn + n0 + bq * 4], 0, 1);
        }
        __syncthreads();

#pragma unroll
        for (int kf = 0; kf < 2; ++kf) {
            Frag af[2], bfr[4];
#pragma unroll
            for (int mt = 0; mt < 2; ++mt)
                load_a(&sa[(moff + mt * 16) * 64], 64, kf * 32, lane, af[mt]);
#pragma unroll
            for (int nt = 0; nt < 4; ++nt)
                load_b(&sb[(noff + nt * 16) * 64], 64, kf * 32, lane, bfr[nt]);
#pragma unroll
            for (int mt = 0; mt < 2; ++mt)
#pragma unroll
                for (int nt = 0; nt < 4; ++nt)
                    acc[mt][nt] = wmma_bf16(af[mt], bfr[nt], acc[mt][nt]);
        }
    }

    // ---- epilogue: bias (+residual) (+relu) ----
#pragma unroll
    for (int mt = 0; mt < 2; ++mt)
#pragma unroll
        for (int nt = 0; nt < 4; ++nt) {
            const int col = n0 + noff + nt * 16 + (lane & 15);
            const float bv = bias[col];
#pragma unroll
            for (int r = 0; r < 8; ++r) {
                const int row = m0 + moff + mt * 16 + r + 8 * (lane >> 4);
                float v = acc[mt][nt][r] + bv;
                if (resid) v += resid[(size_t)row * Nn + col];
                if (relu) v = fmaxf(v, 0.0f);
                Cout[(size_t)row * Nn + col] = v;
            }
        }
}

// ---------------------------------------------------------------------------
// LayerNorm over E=1024, one row per 256-thread block.
// ---------------------------------------------------------------------------
__global__ __launch_bounds__(256) void ln_kernel(const float* __restrict__ X,
                                                 const float* __restrict__ g,
                                                 const float* __restrict__ b,
                                                 float* __restrict__ Y) {
    __shared__ float rs[8], rq[8];
    const int row = blockIdx.x, tid = threadIdx.x, lane = tid & 31, w = tid >> 5;
    const float4 v = ((const float4*)(X + (size_t)row * 1024))[tid];
    float s = v.x + v.y + v.z + v.w;
    float q = v.x * v.x + v.y * v.y + v.z * v.z + v.w * v.w;
#pragma unroll
    for (int m = 1; m < 32; m <<= 1) {
        s += __shfl_xor(s, m, 32);
        q += __shfl_xor(q, m, 32);
    }
    if (lane == 0) { rs[w] = s; rq[w] = q; }
    __syncthreads();
    float ts = 0.0f, tq = 0.0f;
#pragma unroll
    for (int i = 0; i < 8; ++i) { ts += rs[i]; tq += rq[i]; }
    const float mu  = ts * (1.0f / 1024.0f);
    const float var = tq * (1.0f / 1024.0f) - mu * mu;
    const float inv = rsqrtf(var + 1e-5f);
    const float4 gg = ((const float4*)g)[tid];
    const float4 bb = ((const float4*)b)[tid];
    float4 o;
    o.x = (v.x - mu) * inv * gg.x + bb.x;
    o.y = (v.y - mu) * inv * gg.y + bb.y;
    o.z = (v.z - mu) * inv * gg.z + bb.z;
    o.w = (v.w - mu) * inv * gg.w + bb.w;
    ((float4*)(Y + (size_t)row * 1024))[tid] = o;
}

// ---------------------------------------------------------------------------
extern "C" void kernel_launch(void* const* d_in, const int* in_sizes, int n_in,
                              void* d_out, int out_size, void* d_ws, size_t ws_size,
                              hipStream_t stream) {
    const float* value = (const float*)d_in[0];
    const float* key   = (const float*)d_in[1];
    const float* query = (const float*)d_in[2];
    const float* w_out = (const float*)d_in[3];
    const float* b_out = (const float*)d_in[4];
    const float* w1    = (const float*)d_in[5];
    const float* b1    = (const float*)d_in[6];
    const float* w2    = (const float*)d_in[7];
    const float* b2    = (const float*)d_in[8];
    const float* g1    = (const float*)d_in[9];
    const float* be1   = (const float*)d_in[10];
    const float* g2    = (const float*)d_in[11];
    const float* be2   = (const float*)d_in[12];
    float* out = (float*)d_out;
    float* ws  = (float*)d_ws;

    float* attnout = ws;                      // 4096*1024
    float* y0      = ws + 4194304;            // 4096*1024
    float* xb      = ws + 8388608;            // 4096*1024
    float* hid     = ws + 12582912;           // 4096*4096
    float* y1      = attnout;                 // reuse

    // 1) attention -> attnout
    attn_kernel<<<dim3(32, 32), 128, 0, stream>>>(query, key, value, attnout);
    // 2) out-proj + b_out + query residual -> y0
    gemm_kernel<<<dim3(8, 32), 256, 0, stream>>>(attnout, w_out, b_out, query, y0,
                                                 4096, 1024, 1024, 0);
    // 3) LN1 -> xb
    ln_kernel<<<4096, 256, 0, stream>>>(y0, g1, be1, xb);
    // 4) FFN1 + relu -> hid
    gemm_kernel<<<dim3(32, 32), 256, 0, stream>>>(xb, w1, b1, nullptr, hid,
                                                  4096, 4096, 1024, 1);
    // 5) FFN2 + xb residual -> y1
    gemm_kernel<<<dim3(8, 32), 256, 0, stream>>>(hid, w2, b2, xb, y1,
                                                 4096, 1024, 4096, 0);
    // 6) LN2 -> out
    ln_kernel<<<4096, 256, 0, stream>>>(y1, g2, be2, out);
}
